// DGCNN_20486994002748
// MI455X (gfx1250) — matrix operand order repaired and checked by reading, not compiled, not run on hardware
//
#include <hip/hip_runtime.h>
#include <hip/hip_bf16.h>

typedef __attribute__((ext_vector_type(16))) _Float16 v16h;
typedef __attribute__((ext_vector_type(8)))  _Float16 v8h;
typedef __attribute__((ext_vector_type(4)))  _Float16 v4h;
typedef __attribute__((ext_vector_type(8)))  float    v8f;
typedef __attribute__((ext_vector_type(2)))  float    v2f;
typedef __attribute__((ext_vector_type(4)))  int      v4i;

#define B_      16
#define N_PTS   2048
#define KNN     30
#define EDGE_BLOCKS (B_ * N_PTS / 8)     // 8 waves per block, one node per wave
#define KNN_TILE 16                      // node rows per kNN block
#define KNN_LDS_BYTES (KNN_TILE * N_PTS * 4)

// CDNA5 async global->LDS copy path (ASYNCcnt), feature-detected so the kernel
// falls back to direct global loads if this toolchain lacks the builtins.
#if __has_builtin(__builtin_amdgcn_global_load_async_to_lds_b128)
#define HAVE_ASYNC 1
#else
#define HAVE_ASYNC 0
#endif

#if HAVE_ASYNC
// signature (probe-confirmed by diagnostics): (v4i AS1*, v4i AS3*, i32, i32)
__device__ inline void async_copy_b128(const float* gsrc, float* ldst) {
    __builtin_amdgcn_global_load_async_to_lds_b128(
        (__attribute__((address_space(1))) v4i*)gsrc,
        (__attribute__((address_space(3))) v4i*)ldst, 0, 0);
}
#if __has_builtin(__builtin_amdgcn_s_wait_asynccnt)
#define WAIT_ASYNC(n) __builtin_amdgcn_s_wait_asynccnt(n)
#else
#define WAIT_ASYNC(n) asm volatile("s_wait_asynccnt " #n ::: "memory")
#endif
#endif

// ---------------------------------------------------------------------------
// WMMA operand loaders (CDNA5 wave32 VGPR layouts, CDNA5 ISA 7.12.2)
// A f16 (16x32): lane<16 -> M=lane,  halves 0..7 = K kb..kb+7, 8..15 = K 16+kb..
//                (kb = 0 for lanes 0-15, 8 for lanes 16-31)
// B f16 (32x16): lane<16 -> N=lane, K order same as A (kb..kb+7, 16+kb..16+kb+7)
//   -> stage weights TRANSPOSED [N][K] so each lane reads two runs of 8 contiguous halves.
// C/D f32 (16x16): vgpr r: lanes 0-15 (M=r, N=lane); lanes 16-31 (M=r+8, N=l-16)
// All fragment loads are 2x 128-bit LDS loads (ds_load_b128), not u16 gathers.
// ---------------------------------------------------------------------------
__device__ inline v16h load_a(const _Float16* src, int ld, int k0, int lane) {
    int m  = lane & 15;
    int kb = (lane >> 4) << 3;           // 0 or 8
    const _Float16* p = src + (size_t)m * ld + k0 + kb;
    v8h lo = *(const v8h*)(p);           // K kb .. kb+7   (16B aligned)
    v8h hi = *(const v8h*)(p + 16);      // K 16+kb .. 16+kb+7
    return __builtin_shufflevector(lo, hi, 0, 1, 2, 3, 4, 5, 6, 7,
                                   8, 9, 10, 11, 12, 13, 14, 15);
}

__device__ inline v16h load_bt(const _Float16* srcT, int ldk, int k0, int n0, int lane) {
    int n  = n0 + (lane & 15);
    int kb = (lane >> 4) << 3;           // 0 or 8
    const _Float16* p = srcT + (size_t)n * ldk + k0 + kb;
    v8h lo = *(const v8h*)(p);
    v8h hi = *(const v8h*)(p + 16);
    return __builtin_shufflevector(lo, hi, 0, 1, 2, 3, 4, 5, 6, 7,
                                   8, 9, 10, 11, 12, 13, 14, 15);
}

// 32x32 = A[32 x KDIM] @ B[KDIM x 32] + bias(col). Bt is transposed [32][ldB].
template <int KDIM>
__device__ inline void gemm32(const _Float16* A, int ldA,
                              const _Float16* Bt, int ldB,
                              const float* __restrict__ bias,
                              v8f (&out)[2][2], int lane) {
    constexpr int NK = KDIM / 32;
    v16h a[2][NK];
#pragma unroll
    for (int mt = 0; mt < 2; ++mt)
#pragma unroll
        for (int ks = 0; ks < NK; ++ks)
            a[mt][ks] = load_a(A + (size_t)(mt * 16) * ldA, ldA, ks * 32, lane);
    int ncol_lo = lane & 15;
#pragma unroll
    for (int nt = 0; nt < 2; ++nt) {
        v16h bt[NK];
#pragma unroll
        for (int ks = 0; ks < NK; ++ks) bt[ks] = load_bt(Bt, ldB, ks * 32, nt * 16, lane);
        float bc = bias[nt * 16 + ncol_lo];
#pragma unroll
        for (int mt = 0; mt < 2; ++mt) {
            v8f acc = {};
#pragma unroll
            for (int ks = 0; ks < NK; ++ks)
                acc = __builtin_amdgcn_wmma_f32_16x16x32_f16(
                    false, a[mt][ks], false, bt[ks], (short)0, acc, false, false);
#pragma unroll
            for (int r = 0; r < 8; ++r) acc[r] += bc;
            out[mt][nt] = acc;
        }
    }
}

// Accumulate per-feature sum / sumsq over valid edge rows (M < KNN)
__device__ inline void stats_accum(const v8f (&h)[2][2], int lane, float* sAcc) {
    int ncol_lo = lane & 15;
    int off = (lane >> 4) << 3;
#pragma unroll
    for (int nt = 0; nt < 2; ++nt) {
        float s = 0.f, q = 0.f;
#pragma unroll
        for (int mt = 0; mt < 2; ++mt)
#pragma unroll
            for (int r = 0; r < 8; ++r) {
                int m = mt * 16 + r + off;
                if (m < KNN) { float v = h[mt][nt][r]; s += v; q += v * v; }
            }
        s += __shfl_xor(s, 16, 32);
        q += __shfl_xor(q, 16, 32);
        if (lane < 16) {
            atomicAdd(&sAcc[nt * 16 + ncol_lo], s);
            atomicAdd(&sAcc[32 + nt * 16 + ncol_lo], q);
        }
    }
}

// y = relu(h*scale + shift) stored to f16 LDS tile [32][32] (A layout for next GEMM)
__device__ inline void bnrelu_store(const v8f (&h)[2][2], const float* __restrict__ ss,
                                    _Float16 (*sAw)[32], int lane) {
    int ncol_lo = lane & 15;
    int off = (lane >> 4) << 3;
#pragma unroll
    for (int nt = 0; nt < 2; ++nt) {
        int n = nt * 16 + ncol_lo;
        float sc = ss[n], sh = ss[32 + n];
#pragma unroll
        for (int mt = 0; mt < 2; ++mt)
#pragma unroll
            for (int r = 0; r < 8; ++r) {
                int m = mt * 16 + r + off;
                float v = h[mt][nt][r] * sc + sh;
                sAw[m][n] = (_Float16)(v > 0.f ? v : 0.f);
            }
    }
}

// max over valid rows per column, write 32 floats
__device__ inline void max_store(const v8f (&h)[2][2], float* __restrict__ out32, int lane) {
    int ncol_lo = lane & 15;
    int off = (lane >> 4) << 3;
#pragma unroll
    for (int nt = 0; nt < 2; ++nt) {
        float mx = -3.0e38f;
#pragma unroll
        for (int mt = 0; mt < 2; ++mt)
#pragma unroll
            for (int r = 0; r < 8; ++r) {
                int m = mt * 16 + r + off;
                if (m < KNN) mx = fmaxf(mx, h[mt][nt][r]);
            }
        mx = fmaxf(mx, __shfl_xor(mx, 16, 32));
        if (lane < 16) out32[nt * 16 + ncol_lo] = mx;
    }
}

__device__ inline void cvt_store4(_Float16* dst, float4 v) {
    v4h h; h[0] = (_Float16)v.x; h[1] = (_Float16)v.y;
    h[2] = (_Float16)v.z; h[3] = (_Float16)v.w;
    *(v4h*)dst = h;
}

// ---------------------------------------------------------------------------
// kNN via fp32 WMMA Gram matrix: d = n2_i + n2_j - 2 * (x_i . x_j) (reference
// formula, fp32, ranking-safe). One block per 16-node tile; each wave builds
// 16x16 Gram tiles with v_wmma_f32_16x16x4_f32 (x_j chunks async-staged into
// LDS when the toolchain exposes the ASYNCcnt path), writes ordered uint keys
// into a 16 x 2048 dynamic-LDS distance tile, then does wave-per-row 30-way
// argmin selection (ties -> lower index, like jax.lax.top_k).
// ---------------------------------------------------------------------------
template <int FIN>
__global__ __launch_bounds__(256) void knn_kernel(const float* __restrict__ x,
                                                  int* __restrict__ idx) {
    extern __shared__ unsigned sdyn[];
    unsigned (*sdu)[N_PTS] = reinterpret_cast<unsigned (*)[N_PTS]>(sdyn);
    __shared__ float sxi[KNN_TILE][32];
    __shared__ float sn2i[KNN_TILE];
#if HAVE_ASYNC
    __shared__ float sXj[(FIN == 32) ? 8 : 1][2][16][32];  // double-buffered x_j chunk
#endif

    const int tilesPerB = N_PTS / KNN_TILE;
    int b  = blockIdx.x / tilesPerB;
    int i0 = (blockIdx.x - b * tilesPerB) * KNN_TILE;
    int t = threadIdx.x;

    for (int p = t; p < KNN_TILE * 32; p += 256) {
        int r = p >> 5, c = p & 31;
        sxi[r][c] = (c < FIN) ? x[((size_t)b * N_PTS + i0 + r) * FIN + c] : 0.f;
    }
    __syncthreads();
    if (t < KNN_TILE) {
        float s = 0.f;
#pragma unroll
        for (int c = 0; c < FIN; ++c) { float v = sxi[t][c]; s += v * v; }
        sn2i[t] = s;
    }
    __syncthreads();

    int wave = t >> 5, lane = t & 31;
    int mn   = lane & 15;
    int off8 = (lane >> 4) << 3;
    int kk   = (lane >> 4) << 1;         // A/B hold K = kk, kk+1 per 4-wide step
    float n2ir[8];
#pragma unroll
    for (int r = 0; r < 8; ++r) n2ir[r] = sn2i[r + off8];

    const int NCH = N_PTS / 16;          // 128 j-chunks
#if HAVE_ASYNC
    if (FIN == 32) {
        // each async b128 moves 16B/lane: 4 instructions cover a 16x32 f32 chunk
        int arow = (lane >> 3);          // 0..3 within an instruction
        int acol = (lane & 7) * 4;
        int parity = 0;
        {
            int jbase = wave * 16;
#pragma unroll
            for (int q = 0; q < 4; ++q)
                async_copy_b128(x + ((size_t)b * N_PTS + jbase + q * 4 + arow) * 32 + acol,
                                &sXj[wave][0][q * 4 + arow][acol]);
        }
        for (int jc = wave; jc < NCH; jc += 8) {
            int nxt = jc + 8;
            if (nxt < NCH) {             // prefetch next chunk into other buffer
                int jbase = nxt * 16;
#pragma unroll
                for (int q = 0; q < 4; ++q)
                    async_copy_b128(x + ((size_t)b * N_PTS + jbase + q * 4 + arow) * 32 + acol,
                                    &sXj[wave][parity ^ 1][q * 4 + arow][acol]);
                WAIT_ASYNC(4);           // older batch complete, newest in flight
            } else {
                WAIT_ASYNC(0);
            }
            asm volatile("" ::: "memory");
            const float* xjb = &sXj[wave][parity][0][0];
            v8f acc = {};
            float nb2 = 0.f;
#pragma unroll
            for (int ks = 0; ks < 8; ++ks) {
                v2f av = *(const v2f*)(&sxi[mn][ks * 4 + kk]);
                v2f bv = *(const v2f*)(xjb + mn * 32 + ks * 4 + kk);
                nb2 += bv[0] * bv[0] + bv[1] * bv[1];
                acc = __builtin_amdgcn_wmma_f32_16x16x4_f32(
                    false, av, false, bv, (short)0, acc, false, false);
            }
            float n2j = nb2 + __shfl_xor(nb2, 16, 32);
            int jbase = jc * 16;
#pragma unroll
            for (int r = 0; r < 8; ++r) {
                float d = n2ir[r] + n2j - 2.f * acc[r];
                unsigned u = __float_as_uint(d);
                u = (u & 0x80000000u) ? ~u : (u | 0x80000000u);
                sdu[r + off8][jbase + mn] = u;
            }
            parity ^= 1;
        }
    } else
#endif
    {
        for (int jc = wave; jc < NCH; jc += 8) {
            int jbase = jc * 16;
            const float* xrow = x + ((size_t)b * N_PTS + jbase + mn) * FIN;
            v8f acc = {};
            float nb2 = 0.f;
            if (FIN == 32) {
#pragma unroll
                for (int ks = 0; ks < 8; ++ks) {
                    v2f av = *(const v2f*)(&sxi[mn][ks * 4 + kk]);
                    v2f bv = *(const v2f*)(xrow + ks * 4 + kk);
                    nb2 += bv[0] * bv[0] + bv[1] * bv[1];
                    acc = __builtin_amdgcn_wmma_f32_16x16x4_f32(
                        false, av, false, bv, (short)0, acc, false, false);
                }
            } else {
                v2f av = *(const v2f*)(&sxi[mn][kk]);   // K-padded with zeros
                v2f bv;
                bv[0] = (kk < FIN) ? xrow[kk] : 0.f;
                bv[1] = (kk + 1 < FIN) ? xrow[kk + 1] : 0.f;
                nb2 = bv[0] * bv[0] + bv[1] * bv[1];
                acc = __builtin_amdgcn_wmma_f32_16x16x4_f32(
                    false, av, false, bv, (short)0, acc, false, false);
            }
            float n2j = nb2 + __shfl_xor(nb2, 16, 32);
#pragma unroll
            for (int r = 0; r < 8; ++r) {
                float d = n2ir[r] + n2j - 2.f * acc[r];
                unsigned u = __float_as_uint(d);
                u = (u & 0x80000000u) ? ~u : (u | 0x80000000u);
                sdu[r + off8][jbase + mn] = u;
            }
        }
    }
    __syncthreads();

    // selection: wave w owns rows w and w+8; scan with ds_load_b64 (uint2)
    for (int mrow = wave; mrow < KNN_TILE; mrow += 8) {
        size_t orow = ((size_t)b * N_PTS + i0 + mrow) * KNN;
        for (int it = 0; it < KNN; ++it) {
            unsigned long long best = ~0ull;
            for (int j0 = lane * 2; j0 < N_PTS; j0 += 64) {
                uint2 kv = *(const uint2*)(&sdu[mrow][j0]);
                unsigned long long k0 = ((unsigned long long)kv.x << 32) | (unsigned)j0;
                unsigned long long k1 = ((unsigned long long)kv.y << 32) | (unsigned)(j0 + 1);
                if (k0 < best) best = k0;
                if (k1 < best) best = k1;
            }
#pragma unroll
            for (int s = 16; s > 0; s >>= 1) {
                unsigned long long o = __shfl_xor(best, s, 32);
                if (o < best) best = o;
            }
            int jw = (int)(best & 0xffffffffu);
            if (lane == 0) {
                idx[orow + it] = jw;
                sdu[mrow][jw] = 0xffffffffu;     // remove from pool
            }
            asm volatile("s_wait_dscnt 0" ::: "memory");
        }
    }
}

// ---------------------------------------------------------------------------
// Edge MLP, one node per wave (8 nodes / block); GEMMs via v_wmma_f32_16x16x32_f16.
// STAGE 0: stats(h1) -> partials ; STAGE 1: stats(h2) ; STAGE 2: full -> xout
// ---------------------------------------------------------------------------
template <int STAGE, int FIN>
__global__ __launch_bounds__(256) void edge_mlp_kernel(
    const float* __restrict__ x, const int* __restrict__ idx,
    const float* __restrict__ W1, const float* __restrict__ b1,
    const float* __restrict__ W2, const float* __restrict__ b2,
    const float* __restrict__ W3, const float* __restrict__ b3,
    const float* __restrict__ ss1, const float* __restrict__ ss2,
    float* __restrict__ out) {
    __shared__ _Float16 sW1t[32][64];   // transposed [n][k], K zero-padded
    __shared__ _Float16 sW2t[32][32];
    __shared__ _Float16 sW3t[32][32];
    __shared__ _Float16 sE[8][32][64];
    __shared__ _Float16 sA[8][32][32];
    __shared__ float sAcc[64];

    int t = threadIdx.x;
    for (int p = t; p < 64 * 32; p += 256) {
        int k = p >> 5, n = p & 31;
        sW1t[n][k] = (_Float16)((k < 2 * FIN) ? W1[k * 32 + n] : 0.f);
    }
    for (int p = t; p < 32 * 32; p += 256) {
        int k = p >> 5, n = p & 31;
        sW2t[n][k] = (_Float16)W2[k * 32 + n];
        sW3t[n][k] = (_Float16)W3[k * 32 + n];
    }
    if (STAGE != 2) { if (t < 64) sAcc[t] = 0.f; }
    __syncthreads();

    int wave = t >> 5, lane = t & 31;
    int gid = blockIdx.x * 8 + wave;
    int b = gid / N_PTS, i = gid - b * N_PTS;
    const float* xi = x + ((size_t)b * N_PTS + i) * FIN;
    const int* irow = idx + (size_t)gid * KNN;

    {   // build edge rows: lane owns row r; e = [x_i, x_j - x_i], K-padded
        int r = lane;
        int j = (r < KNN) ? irow[r] : 0;
        const float* xj = x + ((size_t)b * N_PTS + j) * FIN;
        float fi[FIN], fj[FIN];
#pragma unroll
        for (int c = 0; c < FIN; ++c) { fi[c] = xi[c]; fj[c] = xj[c]; }
#pragma unroll
        for (int c0 = 0; c0 < 64; c0 += 8) {
            v8h tmp;
#pragma unroll
            for (int u = 0; u < 8; ++u) {
                int c = c0 + u;
                float v = 0.f;
                if (r < KNN) {
                    if (c < FIN)          v = fi[c];
                    else if (c < 2 * FIN) v = fj[c - FIN] - fi[c - FIN];
                }
                tmp[u] = (_Float16)v;
            }
            *(v8h*)(&sE[wave][r][c0]) = tmp;     // 128-bit LDS store
        }
    }
    asm volatile("s_wait_dscnt 0" ::: "memory");   // same-wave cross-lane LDS hazard

    constexpr int KD1 = (2 * FIN <= 32) ? 32 : 64;
    v8f h1[2][2];
    gemm32<KD1>(&sE[wave][0][0], 64, &sW1t[0][0], 64, b1, h1, lane);

    if (STAGE == 0) {
        stats_accum(h1, lane, sAcc);
    } else {
        bnrelu_store(h1, ss1, sA[wave], lane);
        asm volatile("s_wait_dscnt 0" ::: "memory");
        v8f h2[2][2];
        gemm32<32>(&sA[wave][0][0], 32, &sW2t[0][0], 32, b2, h2, lane);
        if (STAGE == 1) {
            stats_accum(h2, lane, sAcc);
        } else {
            bnrelu_store(h2, ss2, sA[wave], lane);   // per-wave buffer; DS in-order
            asm volatile("s_wait_dscnt 0" ::: "memory");
            v8f h3[2][2];
            gemm32<32>(&sA[wave][0][0], 32, &sW3t[0][0], 32, b3, h3, lane);
            max_store(h3, out + ((size_t)b * N_PTS + i) * 32, lane);
        }
    }
    if (STAGE != 2) {
        __syncthreads();
        if (t < 64) out[(size_t)blockIdx.x * 64 + t] = sAcc[t];
    }
}

// reduce per-block partials -> fused BN scale/shift: y = h*scale + shift
__global__ __launch_bounds__(32) void stats_finalize_kernel(
    const float* __restrict__ partials, int nblk,
    const float* __restrict__ g, const float* __restrict__ beta,
    float cnt, float* __restrict__ ss) {
    int t = threadIdx.x;   // 32 threads, one per feature
    float s = 0.f, q = 0.f;
    for (int ib = 0; ib < nblk; ++ib) {
        s += partials[(size_t)ib * 64 + t];
        q += partials[(size_t)ib * 64 + 32 + t];
    }
    float mean = s / cnt;
    float var  = q / cnt - mean * mean;
    float inv  = rsqrtf(var + 1e-5f);
    float sc   = g[t] * inv;
    ss[t]      = sc;
    ss[32 + t] = beta[t] - mean * sc;
}

// ---------------------------------------------------------------------------
// lin1 (concat 96 -> 1024) fused with global max pool over the 2048 nodes.
// ---------------------------------------------------------------------------
__global__ __launch_bounds__(256) void lin1_max_kernel(
    const float* __restrict__ x1, const float* __restrict__ x2,
    const float* __restrict__ x3, const float* __restrict__ W,
    const float* __restrict__ bias, float* __restrict__ y) {
    __shared__ _Float16 sWt[64][96];    // transposed [n][k]
    __shared__ _Float16 sXa[8][16][96];
    __shared__ float sMax[8][16];
    int t = threadIdx.x;
    int colbase = blockIdx.x * 64;
    int b = blockIdx.y;
    for (int p = t; p < 96 * 64; p += 256) {
        int c = p & 63, r = p >> 6;
        sWt[c][r] = (_Float16)W[(size_t)r * 1024 + colbase + c];
    }
    __syncthreads();
    int wave = t >> 5, lane = t & 31;
    int nsub = wave & 3, rh = wave >> 2;
    float rmax[8];
#pragma unroll
    for (int r = 0; r < 8; ++r) rmax[r] = -3.0e38f;

    for (int c = rh; c < N_PTS / 16; c += 2) {
        int m = lane & 15;
        int half = lane >> 4;             // lane halves split the 96 features
        size_t rowbase = ((size_t)b * N_PTS + c * 16 + m) * 32;
        if (c + 2 < N_PTS / 16) {         // global_prefetch_b8 next chunk
            __builtin_prefetch(x1 + rowbase + 16 * 32, 0, 0);
            __builtin_prefetch(x2 + rowbase + 16 * 32, 0, 0);
            __builtin_prefetch(x3 + rowbase + 16 * 32, 0, 0);
        }
        _Float16* dst = &sXa[wave][m][0];
        if (half == 0) {                  // feats 0..47 = x1[0..31] + x2[0..15]
            const float4* p1 = (const float4*)(x1 + rowbase);
            const float4* p2 = (const float4*)(x2 + rowbase);
#pragma unroll
            for (int q = 0; q < 8; ++q) cvt_store4(dst + q * 4, p1[q]);
#pragma unroll
            for (int q = 0; q < 4; ++q) cvt_store4(dst + 32 + q * 4, p2[q]);
        } else {                          // feats 48..95 = x2[16..31] + x3[0..31]
            const float4* p2 = (const float4*)(x2 + rowbase);
            const float4* p3 = (const float4*)(x3 + rowbase);
#pragma unroll
            for (int q = 4; q < 8; ++q) cvt_store4(dst + 32 + q * 4, p2[q]);
#pragma unroll
            for (int q = 0; q < 8; ++q) cvt_store4(dst + 64 + q * 4, p3[q]);
        }
        asm volatile("s_wait_dscnt 0" ::: "memory");
        v8f acc = {};
#pragma unroll
        for (int ks = 0; ks < 3; ++ks) {
            v16h a  = load_a(&sXa[wave][0][0], 96, ks * 32, lane);
            v16h bm = load_bt(&sWt[0][0], 96, ks * 32, nsub * 16, lane);
            acc = __builtin_amdgcn_wmma_f32_16x16x32_f16(
                false, a, false, bm, (short)0, acc, false, false);
        }
#pragma unroll
        for (int r = 0; r < 8; ++r) rmax[r] = fmaxf(rmax[r], acc[r]);
    }
    float mx = rmax[0];
#pragma unroll
    for (int r = 1; r < 8; ++r) mx = fmaxf(mx, rmax[r]);
    mx = fmaxf(mx, __shfl_xor(mx, 16, 32));
    if (lane < 16) sMax[wave][lane] = mx;
    __syncthreads();
    if (wave < 4 && lane < 16) {          // waves w and w+4 share columns
        float tot = fmaxf(sMax[wave][lane], sMax[wave + 4][lane]);
        int col = colbase + wave * 16 + lane;
        y[(size_t)b * 1024 + col] = tot + bias[col];
    }
}

// Output MLP: tiny (16 rows), plain fp32 dot products (float4 on the row).
__global__ __launch_bounds__(256) void linear_kernel(
    const float* __restrict__ in, const float* __restrict__ W,
    const float* __restrict__ bias, float* __restrict__ out,
    int rows, int din, int dout, int dorelu) {
    int gidx = blockIdx.x * blockDim.x + threadIdx.x;
    if (gidx >= rows * dout) return;
    int r = gidx / dout, c = gidx - r * dout;
    const float4* ip4 = (const float4*)(in + (size_t)r * din);
    float acc = bias[c];
    for (int k4 = 0; k4 < (din >> 2); ++k4) {
        float4 v = ip4[k4];
        const float* wp = W + (size_t)(k4 * 4) * dout + c;
        acc += v.x * wp[0] + v.y * wp[(size_t)dout] +
               v.z * wp[(size_t)2 * dout] + v.w * wp[(size_t)3 * dout];
    }
    if (dorelu) acc = fmaxf(acc, 0.f);
    out[gidx] = acc;
}

// ---------------------------------------------------------------------------
template <int FIN>
static void run_conv_layer(const float* xin, float* xout, int* idxb, float* partials,
                           float* ss1, float* ss2,
                           const float* const* Wl, const float* const* bl,
                           const float* const* gl, const float* const* betal,
                           float cnt, hipStream_t stream) {
    knn_kernel<FIN><<<B_ * (N_PTS / KNN_TILE), 256, KNN_LDS_BYTES, stream>>>(xin, idxb);
    edge_mlp_kernel<0, FIN><<<EDGE_BLOCKS, 256, 0, stream>>>(
        xin, idxb, Wl[0], bl[0], Wl[1], bl[1], Wl[2], bl[2], ss1, ss2, partials);
    stats_finalize_kernel<<<1, 32, 0, stream>>>(partials, EDGE_BLOCKS, gl[0], betal[0], cnt, ss1);
    edge_mlp_kernel<1, FIN><<<EDGE_BLOCKS, 256, 0, stream>>>(
        xin, idxb, Wl[0], bl[0], Wl[1], bl[1], Wl[2], bl[2], ss1, ss2, partials);
    stats_finalize_kernel<<<1, 32, 0, stream>>>(partials, EDGE_BLOCKS, gl[1], betal[1], cnt, ss2);
    edge_mlp_kernel<2, FIN><<<EDGE_BLOCKS, 256, 0, stream>>>(
        xin, idxb, Wl[0], bl[0], Wl[1], bl[1], Wl[2], bl[2], ss1, ss2, xout);
}

extern "C" void kernel_launch(void* const* d_in, const int* in_sizes, int n_in,
                              void* d_out, int out_size, void* d_ws, size_t ws_size,
                              hipStream_t stream) {
    (void)in_sizes; (void)n_in; (void)out_size; (void)ws_size;
    const float* data = (const float*)d_in[0];
    const float* cW[3][3]; const float* cb[3][3];
    const float* cg[3][2]; const float* cbeta[3][2];
    for (int l = 0; l < 3; ++l) {
        int base = 1 + l * 10;   // per layer: W0,W1,W2,b0,b1,b2,g0,g1,beta0,beta1
        for (int i = 0; i < 3; ++i) {
            cW[l][i] = (const float*)d_in[base + i];
            cb[l][i] = (const float*)d_in[base + 3 + i];
        }
        for (int i = 0; i < 2; ++i) {
            cg[l][i]    = (const float*)d_in[base + 6 + i];
            cbeta[l][i] = (const float*)d_in[base + 8 + i];
        }
    }
    const float* lin1W = (const float*)d_in[31];
    const float* lin1b = (const float*)d_in[32];
    const float* oW[4]; const float* ob[4];
    for (int i = 0; i < 4; ++i) {
        oW[i] = (const float*)d_in[33 + 2 * i];
        ob[i] = (const float*)d_in[34 + 2 * i];
    }

    // allow 128KB dynamic LDS for the kNN distance tile (320KB/WGP on CDNA5);
    // hipFuncSetAttribute is not stream-ordered -> graph-capture safe.
    (void)hipFuncSetAttribute(reinterpret_cast<const void*>(&knn_kernel<3>),
                              hipFuncAttributeMaxDynamicSharedMemorySize, KNN_LDS_BYTES);
    (void)hipFuncSetAttribute(reinterpret_cast<const void*>(&knn_kernel<32>),
                              hipFuncAttributeMaxDynamicSharedMemorySize, KNN_LDS_BYTES);

    char* ws = (char*)d_ws;
    size_t off = 0;
    auto alloc = [&](size_t bytes) -> void* {
        void* p = (void*)(ws + off);
        off = (off + bytes + 255) & ~(size_t)255;
        return p;
    };
    float* xb[3];
    for (int l = 0; l < 3; ++l) xb[l] = (float*)alloc((size_t)B_ * N_PTS * 32 * 4);
    int*   idxb     = (int*)alloc((size_t)B_ * N_PTS * KNN * 4);
    float* partials = (float*)alloc((size_t)EDGE_BLOCKS * 64 * 4);
    float* ss1      = (float*)alloc(64 * 4);
    float* ss2      = (float*)alloc(64 * 4);
    float* ylin     = (float*)alloc(16 * 1024 * 4);
    float* tb1      = (float*)alloc(16 * 1024 * 4);
    float* tb2      = (float*)alloc(16 * 256 * 4);
    float* tb3      = (float*)alloc(16 * 128 * 4);

    const float cnt = (float)((size_t)B_ * N_PTS * KNN);

    run_conv_layer<3>(data, xb[0], idxb, partials, ss1, ss2,
                      cW[0], cb[0], cg[0], cbeta[0], cnt, stream);
    run_conv_layer<32>(xb[0], xb[1], idxb, partials, ss1, ss2,
                       cW[1], cb[1], cg[1], cbeta[1], cnt, stream);
    run_conv_layer<32>(xb[1], xb[2], idxb, partials, ss1, ss2,
                       cW[2], cb[2], cg[2], cbeta[2], cnt, stream);

    lin1_max_kernel<<<dim3(16, B_), 256, 0, stream>>>(xb[0], xb[1], xb[2], lin1W, lin1b, ylin);

    linear_kernel<<<(16 * 1024 + 255) / 256, 256, 0, stream>>>(ylin, oW[0], ob[0], tb1, 16, 1024, 1024, 1);
    linear_kernel<<<(16 * 256  + 255) / 256, 256, 0, stream>>>(tb1,  oW[1], ob[1], tb2, 16, 1024, 256, 1);
    linear_kernel<<<(16 * 128  + 255) / 256, 256, 0, stream>>>(tb2,  oW[2], ob[2], tb3, 16, 256, 128, 1);
    linear_kernel<<<(16 * 40   + 255) / 256, 256, 0, stream>>>(tb3,  oW[3], ob[3], (float*)d_out, 16, 128, 40, 0);
}
